// Qwen3NextGatedDeltaNet_26886495272969
// MI455X (gfx1250) — matrix operand order repaired and checked by reading, not compiled, not run on hardware
//
#include <hip/hip_runtime.h>
#include <hip/hip_bf16.h>
#include <cstdint>

// ---------------------------------------------------------------------------
// Qwen3-Next Gated DeltaNet forward for MI455X (gfx1250, wave32, WMMA).
// WMMA operands stored K-contiguous (ds_load_b128 fragments); chunk staging
// uses gfx1250 global_load_async_to_lds_b128 + s_wait_asynccnt.
// ---------------------------------------------------------------------------

typedef __attribute__((ext_vector_type(16))) _Float16 v16h;
typedef __attribute__((ext_vector_type(8)))  _Float16 v8h;
typedef __attribute__((ext_vector_type(4)))  _Float16 v4h;
typedef __attribute__((ext_vector_type(8)))  float    v8f;

__device__ __forceinline__ v8f wmma_f32_f16(v16h a, v16h b, v8f c) {
  return __builtin_amdgcn_wmma_f32_16x16x32_f16(
      false, a, false, b, (short)0, c, false, false);
}

// Async DMA: global (f16 data) -> LDS, 16 bytes per lane. LDS address is the
// low 32 bits of the flat pointer (LDS aperture truncates to addr[31:0]).
__device__ __forceinline__ void async_copy_b128(void* lds, const void* gptr) {
  uint32_t l = (uint32_t)(uintptr_t)lds;
  asm volatile("global_load_async_to_lds_b128 %0, %1, off"
               :: "v"(l), "v"(gptr) : "memory");
}
__device__ __forceinline__ void wait_async0() {
  asm volatile("s_wait_asynccnt 0" ::: "memory");
}

// A fragment (16x32 f16): base is [m][k] row-major, ld in halves.
__device__ __forceinline__ v16h frag_a_f16(const _Float16* base, int ld) {
  const int lane = threadIdx.x & 31;
  const _Float16* p = base + (lane & 15) * ld + ((lane >> 4) * 8);
  v8h lo = *(const v8h*)p;
  v8h hi = *(const v8h*)(p + 16);
  return __builtin_shufflevector(lo, hi, 0, 1, 2, 3, 4, 5, 6, 7,
                                 8, 9, 10, 11, 12, 13, 14, 15);
}
// B fragment (32x16 f16): baseT is [n][k] (transposed) layout, ld in halves.
__device__ __forceinline__ v16h frag_b_f16(const _Float16* baseT, int ld) {
  const int lane = threadIdx.x & 31;
  const _Float16* p = baseT + (lane & 15) * ld + ((lane >> 4) * 8);
  v8h lo = *(const v8h*)p;
  v8h hi = *(const v8h*)(p + 16);
  return __builtin_shufflevector(lo, hi, 0, 1, 2, 3, 4, 5, 6, 7,
                                 8, 9, 10, 11, 12, 13, 14, 15);
}
// C/D (16x16 f32, 8 VGPRs): row = r + 8*(lane>>4), col = lane&15.

// ------------------------------- constants ---------------------------------
constexpr int BB = 2, SS = 4096, EE = 2048;
constexpr int HVc = 16, DKc = 128, DVc = 128;
constexpr int KEY_DIM = 1024;
constexpr int VALUE_DIM = 2048;
constexpr int QKVZ_N = 6144;
constexpr int CONV_DIM = 4096;
constexpr int ROWS = BB * SS;  // 8192
constexpr int CHUNKN = 64;

// ---------------------------------------------------------------------------
// f32-in / f32-out GEMM via f16 WMMA. Block = 256 thr = 8 waves in 2x4 grid;
// wave tile 32x32 (2x2 WMMA tiles); block tile 64(M) x 128(N); k-step 32.
// ---------------------------------------------------------------------------
__global__ void wmma_gemm_kernel(const float* __restrict__ A,
                                 const float* __restrict__ Bm,
                                 float* __restrict__ C,
                                 int M, int N, int K) {
  __shared__ _Float16 As[64 * 32];    // [m][k]
  __shared__ _Float16 BsT[128 * 32];  // [n][k]
  const int tid = threadIdx.x;
  const int w = tid >> 5, lane = tid & 31;
  const int wm = w >> 2, wn = w & 3;
  const int m0 = blockIdx.y * 64, n0 = blockIdx.x * 128;
  v8f acc[2][2] = {{{}, {}}, {{}, {}}};

  for (int k0 = 0; k0 < K; k0 += 32) {
#pragma unroll
    for (int v = tid; v < 512; v += 256) {  // As: 64x32 f32 -> f16
      int r = v >> 3, c4 = (v & 7) * 4;
      float4 f = *(const float4*)&A[(size_t)(m0 + r) * K + k0 + c4];
      *(v4h*)&As[r * 32 + c4] =
          (v4h){(_Float16)f.x, (_Float16)f.y, (_Float16)f.z, (_Float16)f.w};
    }
#pragma unroll
    for (int v = tid; v < 1024; v += 256) {  // BsT: transpose 32x128 -> [n][k]
      int kr = v >> 5, n4 = (v & 31) * 4;
      float4 f = *(const float4*)&Bm[(size_t)(k0 + kr) * N + n0 + n4];
      BsT[(n4 + 0) * 32 + kr] = (_Float16)f.x;
      BsT[(n4 + 1) * 32 + kr] = (_Float16)f.y;
      BsT[(n4 + 2) * 32 + kr] = (_Float16)f.z;
      BsT[(n4 + 3) * 32 + kr] = (_Float16)f.w;
    }
    if (k0 + 32 < K) {  // gfx1250 global_prefetch_b8 of next k-tile
      __builtin_prefetch(&A[(size_t)(m0 + (tid >> 2)) * K + (k0 + 32)], 0, 1);
      __builtin_prefetch(&Bm[(size_t)(k0 + 32 + (tid >> 3)) * N + n0], 0, 1);
    }
    __syncthreads();
    v16h aF[2], bF[2];
#pragma unroll
    for (int im = 0; im < 2; ++im)
      aF[im] = frag_a_f16(As + (wm * 32 + im * 16) * 32, 32);
#pragma unroll
    for (int in = 0; in < 2; ++in)
      bF[in] = frag_b_f16(BsT + (wn * 32 + in * 16) * 32, 32);
#pragma unroll
    for (int im = 0; im < 2; ++im)
#pragma unroll
      for (int in = 0; in < 2; ++in)
        acc[im][in] = wmma_f32_f16(aF[im], bF[in], acc[im][in]);
    __syncthreads();
  }

  const int r0 = (lane >> 4) * 8, cn = lane & 15;
#pragma unroll
  for (int im = 0; im < 2; ++im)
#pragma unroll
    for (int in = 0; in < 2; ++in)
#pragma unroll
      for (int r = 0; r < 8; ++r)
        C[(size_t)(m0 + wm * 32 + im * 16 + r + r0) * N +
          (n0 + wn * 32 + in * 16 + cn)] = acc[im][in][r];
}

// ---------------------------------------------------------------------------
// Gate projection: beta = sigmoid(hs@w_b), g = -exp(A)*softplus(hs@w_a + bias)
// ---------------------------------------------------------------------------
__global__ void gate_kernel(const float* __restrict__ hs,
                            const float* __restrict__ w_ba,
                            const float* __restrict__ A_log,
                            const float* __restrict__ dt_bias,
                            float* __restrict__ beta_out,
                            float* __restrict__ g_out) {
  int idx = blockIdx.x * blockDim.x + threadIdx.x;
  if (idx >= ROWS * HVc) return;
  int h = idx & 15;
  int row = idx >> 4;
  const float* x = hs + (size_t)row * EE;
  float accb = 0.f, acca = 0.f;
  for (int e = 0; e < EE; ++e) {
    float xv = x[e];
    accb += xv * w_ba[e * (2 * HVc) + h];
    acca += xv * w_ba[e * (2 * HVc) + HVc + h];
  }
  beta_out[idx] = 1.f / (1.f + expf(-accb));
  float aa = acca + dt_bias[h];
  float sp = (aa > 20.f) ? aa : log1pf(expf(aa));
  g_out[idx] = -expf(A_log[h]) * sp;
}

// ---------------------------------------------------------------------------
// Causal depthwise conv1d (K=4) + SiLU; output stored f16 (halves HBM traffic)
// ---------------------------------------------------------------------------
__global__ void conv_silu_kernel(const float* __restrict__ qkvz,
                                 const float* __restrict__ conv_w,
                                 _Float16* __restrict__ mixed) {
  size_t idx = (size_t)blockIdx.x * 256 + threadIdx.x;
  if (idx >= (size_t)ROWS * CONV_DIM) return;
  int c = (int)(idx & (CONV_DIM - 1));
  size_t bs = idx >> 12;
  int s = (int)(bs & (SS - 1));
  float acc = 0.f;
#pragma unroll
  for (int t = 0; t < 4; ++t) {
    int sp = s - 3 + t;
    if (sp >= 0) acc += conv_w[c * 4 + t] * qkvz[(bs + (size_t)(sp - s)) * QKVZ_N + c];
  }
  mixed[idx] = (_Float16)(acc / (1.f + expf(-acc)));
}

// ---------------------------------------------------------------------------
// Chunked gated delta rule. One WG per (b,h); 8 waves; ~282 KB dynamic LDS.
// Chunk q/k/v tiles staged via async DMA (global_load_async_to_lds_b128).
// ---------------------------------------------------------------------------
__global__ void delta_kernel(const _Float16* __restrict__ mixed,
                             const float* __restrict__ beta_g,
                             const float* __restrict__ g_g,
                             float* __restrict__ core) {
  const int bh = blockIdx.x;
  const int b = bh >> 4, h = bh & 15, hq = h >> 1;  // GQA rep=2
  const int tid = threadIdx.x, w = tid >> 5, lane = tid & 31;
  const int r0 = (lane >> 4) * 8, cn = lane & 15;

  extern __shared__ char smem_raw[];
  float* St   = (float*)smem_raw;       // [dv][dk] f32 state      (64 KB)
  float* Af   = St + 128 * 128;         // [pos][pos] f32 T matrix (16 KB)
  float* rowb = Af + 64 * 64;
  float* gcum = rowb + 64;
  float* graw = gcum + 64;
  float* betaL = graw + 64;
  float* ksc  = betaL + 64;  // beta*exp(gcum)
  float* qsc  = ksc + 64;    // exp(gcum)
  float* kgs  = qsc + 64;    // exp(gcum[63]-gcum)
  _Float16* StH  = (_Float16*)(rowb + 512);  // [dv][dk] f16 mirror (32 KB)
  _Float16* qn   = StH + 128 * 128;          // [pos][d]
  _Float16* kn   = qn + 64 * 128;            // [pos][d]
  _Float16* W2   = kn + 64 * 128;            // k_cumdecay [pos][d]
  _Float16* W1T  = W2 + 64 * 128;            // value_intra/v_new [d][pos]
  _Float16* kbgT = W1T + 128 * 64;           // (k*beta*e^g)^T [d][pos]
  _Float16* kgT  = kbgT + 128 * 64;          // (k*e^{g63-g})^T [d][pos]
  _Float16* vbT  = kgT + 128 * 64;           // (v*beta)^T [d][pos]
  _Float16* Tf16 = vbT + 128 * 64;           // T / attn_i f16 [pos][pos]
  _Float16* qstage = Tf16 + 64 * 64;         // raw q rows [pos][d]
  _Float16* kstage = qstage + 64 * 128;      // raw k rows [pos][d]
  _Float16* vstage = kstage + 64 * 128;      // raw v rows [pos][d]

  for (int i = tid; i < 128 * 128; i += 256) { St[i] = 0.f; StH[i] = (_Float16)0.f; }
  __syncthreads();

  for (int c = 0; c < SS / CHUNKN; ++c) {
    const int sbase = c * CHUNKN;
    // ---- stage 0: async DMA q/k/v chunk tiles into LDS (ASYNCcnt path) ----
    // 3 planes * 64 rows * 16 chunks(16B) = 3072 lane-ops, 12 per thread.
#pragma unroll
    for (int e = tid; e < 3072; e += 256) {
      int plane = e >> 10;          // 0:q 1:k 2:v
      int r = (e >> 4) & 63;
      int ch = e & 15;              // 16-byte chunk within the row
      int s = sbase + r;
      int gcol = (plane == 0) ? (hq * DKc)
               : (plane == 1) ? (KEY_DIM + hq * DKc)
                              : (2 * KEY_DIM + h * DVc);
      const _Float16* g =
          mixed + (size_t)(b * SS + s) * CONV_DIM + gcol + ch * 8;
      _Float16* l = (plane == 0 ? qstage : plane == 1 ? kstage : vstage) +
                    r * 128 + ch * 8;
      async_copy_b128(l, g);
    }
    if (tid < 64) {
      int s = sbase + tid;
      graw[tid] = g_g[(size_t)(b * SS + s) * HVc + h];
      betaL[tid] = beta_g[(size_t)(b * SS + s) * HVc + h];
    }
    wait_async0();   // s_wait_asynccnt 0
    __syncthreads();

    // ---- stage 1: l2norm q/k from staged LDS rows ----
    if (tid < 64) {
      int r = tid;
      float ssum = 1e-6f;
#pragma unroll
      for (int v = 0; v < 16; ++v) {
        v8h x = *(const v8h*)(qstage + r * 128 + v * 8);
#pragma unroll
        for (int u = 0; u < 8; ++u) { float f = (float)x[u]; ssum += f * f; }
      }
      float rs = rsqrtf(ssum) * 0.08838834764831845f;  // * 1/sqrt(Dk)
#pragma unroll
      for (int v = 0; v < 16; ++v) {
        v8h x = *(const v8h*)(qstage + r * 128 + v * 8);
        v8h y;
#pragma unroll
        for (int u = 0; u < 8; ++u) y[u] = (_Float16)((float)x[u] * rs);
        *(v8h*)(qn + r * 128 + v * 8) = y;
      }
    } else if (tid < 128) {
      int r = tid - 64;
      float ssum = 1e-6f;
#pragma unroll
      for (int v = 0; v < 16; ++v) {
        v8h x = *(const v8h*)(kstage + r * 128 + v * 8);
#pragma unroll
        for (int u = 0; u < 8; ++u) { float f = (float)x[u]; ssum += f * f; }
      }
      float rs = rsqrtf(ssum);
#pragma unroll
      for (int v = 0; v < 16; ++v) {
        v8h x = *(const v8h*)(kstage + r * 128 + v * 8);
        v8h y;
#pragma unroll
        for (int u = 0; u < 8; ++u) y[u] = (_Float16)((float)x[u] * rs);
        *(v8h*)(kn + r * 128 + v * 8) = y;
      }
    }
    __syncthreads();
    if (tid < 64) {
      float acc = 0.f;
      for (int j = 0; j <= tid; ++j) acc += graw[j];
      gcum[tid] = acc;
    }
    __syncthreads();
    if (tid < 64) {
      float gl = gcum[63];
      ksc[tid] = betaL[tid] * expf(gcum[tid]);
      qsc[tid] = expf(gcum[tid]);
      kgs[tid] = expf(gl - gcum[tid]);
    }
    __syncthreads();
    // ---- prescale + transpose pass: kbgT, kgT, vbT (all from LDS) ----
    for (int e = tid; e < 128 * 64; e += 256) {
      int p = e >> 7, d = e & 127;
      float kv = (float)kn[p * 128 + d];
      kbgT[d * 64 + p] = (_Float16)(kv * ksc[p]);
      kgT[d * 64 + p] = (_Float16)(kv * kgs[p]);
      float vv = (float)vstage[p * 128 + d];
      vbT[d * 64 + p] = (_Float16)(vv * betaL[p]);
    }
    __syncthreads();

    // ---- mm1: attn = -(k_beta @ k^T) * decay, strict lower (beta folded) ----
    for (int t = w * 2; t < w * 2 + 2; ++t) {
      int tm = t >> 2, tn = t & 3;
      v8f acc = {};
#pragma unroll
      for (int kk = 0; kk < 4; ++kk) {
        v16h af = frag_a_f16(kn + (tm * 16) * 128 + kk * 32, 128);
        v16h bf = frag_b_f16(kn + (tn * 16) * 128 + kk * 32, 128);
        acc = wmma_f32_f16(af, bf, acc);
      }
#pragma unroll
      for (int r = 0; r < 8; ++r) {
        int i = tm * 16 + r + r0, j = tn * 16 + cn;
        Af[i * 64 + j] = (j < i) ? (-acc[r] * betaL[i] * expf(gcum[i] - gcum[j])) : 0.f;
      }
    }
    __syncthreads();

    // ---- UT transform: sequential forward substitution (f32), then + I ----
    for (int i = 1; i < 64; ++i) {
      if (tid < 64) rowb[tid] = (tid < i) ? Af[i * 64 + tid] : 0.f;
      __syncthreads();
      if (tid < i) {
        float acc = rowb[tid];
        for (int j = 0; j < i; ++j) acc += rowb[j] * Af[j * 64 + tid];
        Af[i * 64 + tid] = acc;
      }
      __syncthreads();
    }
    if (tid < 64) Af[tid * 64 + tid] = 1.0f;
    __syncthreads();
    for (int e = tid; e < 64 * 64; e += 256) Tf16[e] = (_Float16)Af[e];
    __syncthreads();

    // ---- mm2: W1 = T @ v_beta (store ^T) ; W2 = T @ kbg ----
    for (int t = w * 4; t < w * 4 + 4; ++t) {
      int tm = t >> 3, tn = t & 7;
      v8f a1 = {}, a2 = {};
#pragma unroll
      for (int kk = 0; kk < 2; ++kk) {
        v16h af = frag_a_f16(Tf16 + (tm * 16) * 64 + kk * 32, 64);
        v16h b1 = frag_b_f16(vbT + (tn * 16) * 64 + kk * 32, 64);
        v16h b2 = frag_b_f16(kbgT + (tn * 16) * 64 + kk * 32, 64);
        a1 = wmma_f32_f16(af, b1, a1);
        a2 = wmma_f32_f16(af, b2, a2);
      }
#pragma unroll
      for (int r = 0; r < 8; ++r) {
        int row = tm * 16 + r + r0, col = tn * 16 + cn;
        W1T[col * 64 + row] = (_Float16)a1[r];
        W2[row * 128 + col] = (_Float16)a2[r];
      }
    }
    __syncthreads();

    // ---- mm3: v_new = W1 - W2 @ S ; oacc = q @ S (exp(gcum) folded out) ----
    v8f oacc[4];
    for (int i = 0; i < 4; ++i) {
      int t = w * 4 + i, tm = t >> 3, tn = t & 7;
      v8f av = {}, ao = {};
#pragma unroll
      for (int kk = 0; kk < 4; ++kk) {
        v16h bf = frag_b_f16(StH + (tn * 16) * 128 + kk * 32, 128);
        v16h aw = frag_a_f16(W2 + (tm * 16) * 128 + kk * 32, 128);
        v16h aq = frag_a_f16(qn + (tm * 16) * 128 + kk * 32, 128);
        av = wmma_f32_f16(aw, bf, av);
        ao = wmma_f32_f16(aq, bf, ao);
      }
#pragma unroll
      for (int r = 0; r < 8; ++r) {
        int row = tm * 16 + r + r0, col = tn * 16 + cn;
        float vnew = (float)W1T[col * 64 + row] - av[r];
        W1T[col * 64 + row] = (_Float16)vnew;
        ao[r] *= qsc[row];
      }
      oacc[i] = ao;
    }
    __syncthreads();

    // ---- mm4: attn_i = (q @ k^T) * decay, lower incl. diagonal -> Tf16 ----
    for (int t = w * 2; t < w * 2 + 2; ++t) {
      int tm = t >> 2, tn = t & 3;
      v8f acc = {};
#pragma unroll
      for (int kk = 0; kk < 4; ++kk) {
        v16h af = frag_a_f16(qn + (tm * 16) * 128 + kk * 32, 128);
        v16h bf = frag_b_f16(kn + (tn * 16) * 128 + kk * 32, 128);
        acc = wmma_f32_f16(af, bf, acc);
      }
#pragma unroll
      for (int r = 0; r < 8; ++r) {
        int i = tm * 16 + r + r0, j = tn * 16 + cn;
        Tf16[i * 64 + j] =
            (_Float16)((j <= i) ? (acc[r] * expf(gcum[i] - gcum[j])) : 0.f);
      }
    }
    __syncthreads();

    // ---- mm5: out = oacc + attn_i @ v_new ; store to core ----
    for (int i = 0; i < 4; ++i) {
      int t = w * 4 + i, tm = t >> 3, tn = t & 7;
      v8f acc = oacc[i];
#pragma unroll
      for (int kk = 0; kk < 2; ++kk) {
        v16h af = frag_a_f16(Tf16 + (tm * 16) * 64 + kk * 32, 64);
        v16h bf = frag_b_f16(W1T + (tn * 16) * 64 + kk * 32, 64);
        acc = wmma_f32_f16(af, bf, acc);
      }
#pragma unroll
      for (int r = 0; r < 8; ++r) {
        int row = tm * 16 + r + r0, col = tn * 16 + cn;
        int s = sbase + row;
        core[((size_t)(b * SS + s) * HVc + h) * DVc + col] = acc[r];
      }
    }
    __syncthreads();

    // ---- mm6: S = S*exp(gcum[63]) + kg^T @ v_new (St f32 + StH f16) ----
    float eg = expf(gcum[63]);
    for (int t = w * 8; t < w * 8 + 8; ++t) {
      int tm = t >> 3, tn = t & 7;
      v8f acc = {};
#pragma unroll
      for (int kk = 0; kk < 2; ++kk) {
        v16h af = frag_a_f16(kgT + (tm * 16) * 64 + kk * 32, 64);
        v16h bf = frag_b_f16(W1T + (tn * 16) * 64 + kk * 32, 64);
        acc = wmma_f32_f16(af, bf, acc);
      }
#pragma unroll
      for (int r = 0; r < 8; ++r) {
        int row = tm * 16 + r + r0, col = tn * 16 + cn;  // row=dk, col=dv
        int idx = col * 128 + row;
        float nv = St[idx] * eg + acc[r];
        St[idx] = nv;
        StH[idx] = (_Float16)nv;
      }
    }
    __syncthreads();
  }
}

// ---------------------------------------------------------------------------
// Gated RMSNorm per head: gated = core * rsqrt(mean(core^2)+eps) * w * silu(z)
// ---------------------------------------------------------------------------
__global__ void normgate_kernel(const float* __restrict__ core,
                                const float* __restrict__ qkvz,
                                const float* __restrict__ norm_w,
                                float* __restrict__ gated) {
  int row = blockIdx.x;  // (b*S+s)*HV + h
  int d = threadIdx.x;   // 0..127
  int h = row & 15;
  size_t bs = (size_t)(row >> 4);
  float v = core[(size_t)row * DVc + d];
  float ss = v * v;
#pragma unroll
  for (int off = 16; off > 0; off >>= 1) ss += __shfl_xor(ss, off, 32);
  __shared__ float part[4];
  if ((threadIdx.x & 31) == 0) part[threadIdx.x >> 5] = ss;
  __syncthreads();
  float tot = part[0] + part[1] + part[2] + part[3];
  float inv = rsqrtf(tot * (1.0f / DVc) + 1e-6f);
  float z = qkvz[bs * QKVZ_N + CONV_DIM + h * DVc + d];
  float sz = z / (1.f + expf(-z));
  gated[(size_t)row * DVc + d] = v * inv * norm_w[d] * sz;
}

// ---------------------------------------------------------------------------
extern "C" void kernel_launch(void* const* d_in, const int* in_sizes, int n_in,
                              void* d_out, int out_size, void* d_ws, size_t ws_size,
                              hipStream_t stream) {
  const float* hidden  = (const float*)d_in[0];
  const float* w_qkvz  = (const float*)d_in[1];
  const float* w_ba    = (const float*)d_in[2];
  const float* conv_w  = (const float*)d_in[3];
  const float* A_log   = (const float*)d_in[4];
  const float* dt_bias = (const float*)d_in[5];
  const float* norm_w  = (const float*)d_in[6];
  const float* w_out   = (const float*)d_in[7];
  float* out = (float*)d_out;

  float* ws = (float*)d_ws;
  float* qkvz = ws;                                    // ROWS*6144 f32
  float* betab = qkvz + (size_t)ROWS * QKVZ_N;         // ROWS*16
  float* gb = betab + (size_t)ROWS * HVc;              // ROWS*16
  float* corep = gb + (size_t)ROWS * HVc;              // ROWS*2048
  float* gated = corep + (size_t)ROWS * VALUE_DIM;     // ROWS*2048
  _Float16* mixed = (_Float16*)(gated + (size_t)ROWS * VALUE_DIM);  // ROWS*4096 f16

  // A: qkvz projection GEMM (8192 x 2048 x 6144)
  wmma_gemm_kernel<<<dim3(QKVZ_N / 128, ROWS / 64), 256, 0, stream>>>(
      hidden, w_qkvz, qkvz, ROWS, QKVZ_N, EE);

  // C-gates: beta / g
  gate_kernel<<<(ROWS * HVc) / 256, 256, 0, stream>>>(
      hidden, w_ba, A_log, dt_bias, betab, gb);

  // B: causal depthwise conv + SiLU (f16 output)
  conv_silu_kernel<<<(int)(((size_t)ROWS * CONV_DIM) / 256), 256, 0, stream>>>(
      qkvz, conv_w, mixed);

  // C: chunked gated delta rule (~282 KB dynamic LDS per WG)
  constexpr int DELTA_SMEM =
      (128 * 128 + 64 * 64 + 512) * 4 +
      (128 * 128 + 3 * 64 * 128 + 3 * 128 * 64 + 64 * 64 + 3 * 64 * 128) * 2;
  hipFuncSetAttribute(reinterpret_cast<const void*>(delta_kernel),
                      hipFuncAttributeMaxDynamicSharedMemorySize, DELTA_SMEM);
  delta_kernel<<<BB * HVc, 256, DELTA_SMEM, stream>>>(mixed, betab, gb, corep);

  // D1: gated RMSNorm
  normgate_kernel<<<ROWS * HVc, 128, 0, stream>>>(corep, qkvz, norm_w, gated);

  // D2: output projection GEMM (8192 x 2048 x 2048)
  wmma_gemm_kernel<<<dim3(EE / 128, ROWS / 64), 256, 0, stream>>>(
      gated, w_out, out, ROWS, EE, VALUE_DIM);
}
